// NestedAttention_76252849373683
// MI455X (gfx1250) — matrix operand order repaired and checked by reading, not compiled, not run on hardware
//
#include <hip/hip_runtime.h>
#include <hip/hip_bf16.h>

// ---------------------------------------------------------------------------
// Types for CDNA5 WMMA (wave32)
// ---------------------------------------------------------------------------
typedef __attribute__((ext_vector_type(16))) __bf16 v16bf;
typedef __attribute__((ext_vector_type(8)))  __bf16 v8bf;
typedef __attribute__((ext_vector_type(8)))  float  v8f;

__device__ __forceinline__ __bf16 f2bf(float f) {
  unsigned u = __builtin_bit_cast(unsigned, f);
  unsigned r = (u + 0x7FFFu + ((u >> 16) & 1u)) >> 16;
  unsigned short s = (unsigned short)r;
  return __builtin_bit_cast(__bf16, s);
}

__device__ __forceinline__ float bf2f(__bf16 b) {
  unsigned short s = __builtin_bit_cast(unsigned short, b);
  unsigned u = ((unsigned)s) << 16;
  return __builtin_bit_cast(float, u);
}

// A-matrix fragment (16x32 bf16): lane holds row m = lane&15, half hi = lane>>4.
// elements 0..7  -> K = kc + hi*16 + i
// elements 8..15 -> K = kc + hi*16 + 8 + (i-8)
__device__ __forceinline__ v16bf ldA(const __bf16* __restrict__ rowbase, int kc, int hi) {
  v8bf a0 = *(const v8bf*)(rowbase + kc + hi * 16);
  v8bf a1 = *(const v8bf*)(rowbase + kc + hi * 16 + 8);
  v16bf r;
#pragma unroll
  for (int i = 0; i < 8; ++i) { r[i] = a0[i]; r[8 + i] = a1[i]; }
  return r;
}

__device__ __forceinline__ v8f wmma_bf16(v16bf a, v16bf b, v8f c) {
  // (neg_a, A, neg_b, B, c_mod, C, reuse_a, reuse_b)
  return __builtin_amdgcn_wmma_f32_16x16x32_bf16(false, a, false, b, (short)0, c,
                                                 false, false);
}

#define BB 16
#define NN 1024
#define DD 1024
#define HH 16
#define HD 64

// ---------------------------------------------------------------------------
// Kernel 1: nested feature mask + f32->bf16, record d_tok per token
// ---------------------------------------------------------------------------
__global__ __launch_bounds__(256) void na_prep(const float* __restrict__ X,
                                               const int* __restrict__ em,
                                               __bf16* __restrict__ Xb,
                                               int* __restrict__ dtok) {
  const int tkn = blockIdx.x;
  const int e = em[tkn];
  const int dt = DD >> (3 - e);  // E-1-e with E=4
  if (threadIdx.x == 0) dtok[tkn] = dt;
  const float* src = X + (size_t)tkn * DD;
  __bf16* dst = Xb + (size_t)tkn * DD;
  for (int c = threadIdx.x; c < DD; c += 256) {
    float v = (c < dt) ? src[c] : 0.0f;
    dst[c] = f2bf(v);
  }
}

// ---------------------------------------------------------------------------
// Kernel 2: generic f32 -> bf16 weight conversion
// ---------------------------------------------------------------------------
__global__ __launch_bounds__(256) void na_cvt(const float* __restrict__ s,
                                              __bf16* __restrict__ d, int n) {
  int i = blockIdx.x * 256 + threadIdx.x;
  if (i < n) d[i] = f2bf(s[i]);
}

// ---------------------------------------------------------------------------
// Kernel 3: QKV GEMM  qkv[m,f] = sum_k X[m,k]*W[f,k]
// Register-blocked: each wave computes a 16(M) x 64(F) tile -> 1 A fragment
// reused across 4 B fragments / 4 accumulators per k-chunk (4 WMMA per A load).
// Scatter epilogue: Q,K -> [B,H,N,64] row-major; V -> transposed [B,H,64,N]
// grid (M/16, 3D/256), block 128 (4 waves x 64 F-columns each)
// ---------------------------------------------------------------------------
__global__ __launch_bounds__(128) void na_qkv_gemm(const __bf16* __restrict__ X,
                                                   const __bf16* __restrict__ W,
                                                   __bf16* __restrict__ Q,
                                                   __bf16* __restrict__ K,
                                                   __bf16* __restrict__ VT) {
  const int lane = threadIdx.x & 31, wave = threadIdx.x >> 5;
  const int hi = lane >> 4, lo = lane & 15;
  const int m0 = blockIdx.x * 16;
  const int f0 = (blockIdx.y * 4 + wave) * 64;  // 64 F columns per wave
  const __bf16* arow = X + (size_t)(m0 + lo) * DD;
  const __bf16* brow0 = W + (size_t)(f0 + 0  + lo) * DD;
  const __bf16* brow1 = W + (size_t)(f0 + 16 + lo) * DD;
  const __bf16* brow2 = W + (size_t)(f0 + 32 + lo) * DD;
  const __bf16* brow3 = W + (size_t)(f0 + 48 + lo) * DD;
  v8f acc0 = {}, acc1 = {}, acc2 = {}, acc3 = {};
#pragma unroll 2
  for (int kc = 0; kc < DD; kc += 32) {
    v16bf a = ldA(arow, kc, hi);
    v16bf b0 = *(const v16bf*)(brow0 + kc + hi * 16);
    v16bf b1 = *(const v16bf*)(brow1 + kc + hi * 16);
    v16bf b2 = *(const v16bf*)(brow2 + kc + hi * 16);
    v16bf b3 = *(const v16bf*)(brow3 + kc + hi * 16);
    acc0 = wmma_bf16(a, b0, acc0);
    acc1 = wmma_bf16(a, b1, acc1);
    acc2 = wmma_bf16(a, b2, acc2);
    acc3 = wmma_bf16(a, b3, acc3);
  }
  v8f accs[4] = {acc0, acc1, acc2, acc3};
#pragma unroll
  for (int fi = 0; fi < 4; ++fi) {
    const int f = f0 + fi * 16 + lo;
    const int s = f >> 10, r = f & (DD - 1), h = r >> 6, t = r & 63;
#pragma unroll
    for (int j = 0; j < 8; ++j) {
      const int tm = m0 + j + hi * 8;
      const int b_ = tm >> 10, n = tm & (NN - 1);
      const __bf16 v = f2bf(accs[fi][j]);
      if (s == 0)
        Q[(((size_t)(b_ * HH + h)) * NN + n) * HD + t] = v;
      else if (s == 1)
        K[(((size_t)(b_ * HH + h)) * NN + n) * HD + t] = v;
      else
        VT[(((size_t)(b_ * HH + h)) * HD + t) * NN + n] = v;
    }
  }
}

// ---------------------------------------------------------------------------
// Kernel 4: attention, one block per (b*H+h, 16-query tile)
// S tile (16 x 1024) kept in LDS as bf16; softmax normalization folded into
// the epilogue (divide by rowsum).
// ---------------------------------------------------------------------------
__global__ __launch_bounds__(128) void na_attn(const __bf16* __restrict__ Q,
                                               const __bf16* __restrict__ K,
                                               const __bf16* __restrict__ VT,
                                               __bf16* __restrict__ AO) {
  __shared__ alignas(32) __bf16 Sb[16][NN];
  __shared__ float rowsum[16];

  const int lane = threadIdx.x & 31, wave = threadIdx.x >> 5;
  const int hi = lane >> 4, lo = lane & 15;
  const int bh = blockIdx.y;       // b*H + h
  const int n0 = blockIdx.x * 16;  // query-tile base

  const size_t headbase = (size_t)bh * NN * HD;
  const __bf16* qrow = Q + headbase + (size_t)(n0 + lo) * HD;
  const v16bf aq0 = ldA(qrow, 0, hi);
  const v16bf aq1 = ldA(qrow, 32, hi);
  const float sc = 0.125f;  // 1/sqrt(64)

  // ---- Phase 1: S = Q K^T * scale, wave w covers keys [w*256, w*256+256)
  for (int kb = 0; kb < 16; ++kb) {
    const int key0 = wave * 256 + kb * 16;
    const __bf16* krow = K + headbase + (size_t)(key0 + lo) * HD;
    v16bf bk0 = *(const v16bf*)(krow + 0 + hi * 16);
    v16bf bk1 = *(const v16bf*)(krow + 32 + hi * 16);
    v8f c = {};
    c = wmma_bf16(aq0, bk0, c);
    c = wmma_bf16(aq1, bk1, c);
#pragma unroll
    for (int j = 0; j < 8; ++j) Sb[j + hi * 8][key0 + lo] = f2bf(c[j] * sc);
  }
  __syncthreads();

  // ---- Phase 2: row softmax (unnormalized exp + per-row sum)
  {
    const int row = threadIdx.x >> 3, sub = threadIdx.x & 7;
    float m = -3.0e38f;
    for (int c = sub; c < NN; c += 8) m = fmaxf(m, bf2f(Sb[row][c]));
#pragma unroll
    for (int o = 4; o; o >>= 1) m = fmaxf(m, __shfl_xor(m, o, 8));
    float s = 0.0f;
    for (int c = sub; c < NN; c += 8) {
      float e = __expf(bf2f(Sb[row][c]) - m);
      s += e;
      Sb[row][c] = f2bf(e);
    }
#pragma unroll
    for (int o = 4; o; o >>= 1) s += __shfl_xor(s, o, 8);
    if (sub == 0) rowsum[row] = s;
  }
  __syncthreads();

  // ---- Phase 3: O = P V, wave w computes hd columns [w*16, w*16+16)
  const int t0 = wave * 16;
  const __bf16* vrow = VT + (size_t)bh * HD * NN + (size_t)(t0 + lo) * NN;
  v8f acc = {};
#pragma unroll 4
  for (int kc = 0; kc < NN; kc += 32) {
    v8bf p0 = *(const v8bf*)(&Sb[lo][kc + hi * 16]);
    v8bf p1 = *(const v8bf*)(&Sb[lo][kc + hi * 16 + 8]);
    v16bf a;
#pragma unroll
    for (int i = 0; i < 8; ++i) { a[i] = p0[i]; a[8 + i] = p1[i]; }
    v16bf b = *(const v16bf*)(vrow + kc + hi * 16);
    acc = wmma_bf16(a, b, acc);
  }
  const int b_ = bh >> 4, h = bh & 15;
#pragma unroll
  for (int j = 0; j < 8; ++j) {
    const int mr = j + hi * 8;
    const float val = acc[j] / rowsum[mr];
    AO[((size_t)(b_ * NN + n0 + mr)) * DD + h * HD + t0 + lo] = f2bf(val);
  }
}

// ---------------------------------------------------------------------------
// Kernel 5: output projection + bias + nested output mask -> f32 d_out
// Register-blocked 16x64 per wave like the QKV GEMM.
// ---------------------------------------------------------------------------
__global__ __launch_bounds__(128) void na_proj_gemm(const __bf16* __restrict__ X,
                                                    const __bf16* __restrict__ W,
                                                    const float* __restrict__ bias,
                                                    const int* __restrict__ dtok,
                                                    float* __restrict__ out) {
  const int lane = threadIdx.x & 31, wave = threadIdx.x >> 5;
  const int hi = lane >> 4, lo = lane & 15;
  const int m0 = blockIdx.x * 16;
  const int f0 = (blockIdx.y * 4 + wave) * 64;
  const __bf16* arow = X + (size_t)(m0 + lo) * DD;
  const __bf16* brow0 = W + (size_t)(f0 + 0  + lo) * DD;
  const __bf16* brow1 = W + (size_t)(f0 + 16 + lo) * DD;
  const __bf16* brow2 = W + (size_t)(f0 + 32 + lo) * DD;
  const __bf16* brow3 = W + (size_t)(f0 + 48 + lo) * DD;
  v8f acc0 = {}, acc1 = {}, acc2 = {}, acc3 = {};
#pragma unroll 2
  for (int kc = 0; kc < DD; kc += 32) {
    v16bf a = ldA(arow, kc, hi);
    v16bf b0 = *(const v16bf*)(brow0 + kc + hi * 16);
    v16bf b1 = *(const v16bf*)(brow1 + kc + hi * 16);
    v16bf b2 = *(const v16bf*)(brow2 + kc + hi * 16);
    v16bf b3 = *(const v16bf*)(brow3 + kc + hi * 16);
    acc0 = wmma_bf16(a, b0, acc0);
    acc1 = wmma_bf16(a, b1, acc1);
    acc2 = wmma_bf16(a, b2, acc2);
    acc3 = wmma_bf16(a, b3, acc3);
  }
  // Hoist per-row nested-dim bounds (same 16 tokens for all 4 f-subtiles)
  int dts[8];
#pragma unroll
  for (int j = 0; j < 8; ++j) dts[j] = dtok[m0 + j + hi * 8];
  v8f accs[4] = {acc0, acc1, acc2, acc3};
#pragma unroll
  for (int fi = 0; fi < 4; ++fi) {
    const int f = f0 + fi * 16 + lo;
    const float pb = bias[f];
#pragma unroll
    for (int j = 0; j < 8; ++j) {
      const int tm = m0 + j + hi * 8;
      out[(size_t)tm * DD + f] = (f < dts[j]) ? (accs[fi][j] + pb) : 0.0f;
    }
  }
}

// ---------------------------------------------------------------------------
// Host launcher
// ---------------------------------------------------------------------------
extern "C" void kernel_launch(void* const* d_in, const int* in_sizes, int n_in,
                              void* d_out, int out_size, void* d_ws, size_t ws_size,
                              hipStream_t stream) {
  (void)in_sizes; (void)n_in; (void)out_size; (void)ws_size;
  const float* in_tokens = (const float*)d_in[0];
  const int*   em        = (const int*)d_in[1];
  const float* qkv_w     = (const float*)d_in[2];
  const float* proj_w    = (const float*)d_in[3];
  const float* proj_b    = (const float*)d_in[4];
  float* out = (float*)d_out;

  const size_t M = (size_t)BB * NN;  // 16384 tokens
  char* ws = (char*)d_ws;
  size_t off = 0;
  __bf16* xb = (__bf16*)(ws + off); off += M * DD * 2;              // 32 MB (also AO)
  __bf16* qw = (__bf16*)(ws + off); off += (size_t)3 * DD * DD * 2; // 6 MB
  __bf16* pw = (__bf16*)(ws + off); off += (size_t)DD * DD * 2;     // 2 MB
  __bf16* qb = (__bf16*)(ws + off); off += M * DD * 2;              // 32 MB
  __bf16* kb = (__bf16*)(ws + off); off += M * DD * 2;              // 32 MB
  __bf16* vT = (__bf16*)(ws + off); off += M * DD * 2;              // 32 MB
  int*  dtok = (int*)(ws + off);    off += M * 4;
  __bf16* ao = xb;  // attention output reuses x_bf16 (x no longer needed)

  // 1) mask + convert tokens, record d_tok
  na_prep<<<dim3((unsigned)M), 256, 0, stream>>>(in_tokens, em, xb, dtok);
  // 2) weight conversion
  {
    int n1 = 3 * DD * DD;
    na_cvt<<<dim3((n1 + 255) / 256), 256, 0, stream>>>(qkv_w, qw, n1);
    int n2 = DD * DD;
    na_cvt<<<dim3((n2 + 255) / 256), 256, 0, stream>>>(proj_w, pw, n2);
  }
  // 3) QKV GEMM (M x 3D x D), 256 f-columns per block
  na_qkv_gemm<<<dim3((unsigned)(M / 16), (3 * DD) / 256), 128, 0, stream>>>(
      xb, qw, qb, kb, vT);
  // 4) attention per (b,h) and 16-query tile
  na_attn<<<dim3(NN / 16, BB * HH), 128, 0, stream>>>(qb, kb, vT, ao);
  // 5) projection + bias + nested output mask
  na_proj_gemm<<<dim3((unsigned)(M / 16), DD / 256), 128, 0, stream>>>(
      ao, pw, proj_b, dtok, out);
}
